// DN4Layer_15831249453834
// MI455X (gfx1250) — matrix operand order repaired and checked by reading, not compiled, not run on hardware
//
#include <hip/hip_runtime.h>
#include <hip/hip_bf16.h>

// ---------------- problem constants (fixed by setup_inputs) ----------------
constexpr int T    = 4;
constexpr int WAY  = 5;
constexpr int SHOT = 5;
constexpr int QRY  = 15;
constexpr int WQ   = WAY * QRY;     // 75 query images per task
constexpr int C    = 640;           // channels (K dim)
constexpr int HW   = 100;           // 10*10 spatial
constexpr int MPAD = 112;           // 100 rows padded to 7*16
constexpr int NREAL= SHOT * HW;     // 500 support descriptors per way
constexpr int NPAD = 512;           // padded to 32*16
constexpr int NB   = 64;            // N-batch staged in LDS (4 tiles of 16)
constexpr int LDSROW = C + 8;       // 648 bf16 -> 1296B row stride (bank fan-out)
constexpr float EPS = 1e-12f;

typedef __attribute__((ext_vector_type(16))) __bf16 v16bf;
typedef __attribute__((ext_vector_type(8)))  float  v8f;

union Frag { uint4 u[2]; v16bf v; };

static __device__ __forceinline__ unsigned short f32_to_bf16(float f) {
    unsigned int u = __float_as_uint(f);
    u += 0x7FFFu + ((u >> 16) & 1u);     // round-to-nearest-even
    return (unsigned short)(u >> 16);
}

static __device__ __forceinline__ void ins3(float x, float& a, float& b, float& c) {
    // insert x into descending triple (a >= b >= c)
    if (x > c) {
        if (x > b) {
            c = b;
            if (x > a) { b = a; a = x; } else { b = x; }
        } else {
            c = x;
        }
    }
}

// ---------------------------------------------------------------------------
// Kernel 1: normalize query over hw (contiguous 100-float run per (t,q,c)),
// convert to bf16, write A-layout Qbf[t*75+q][x(0..111)][c]; rows 100..111 = 0.
// One wave32 per (t,q,c) vector.
// ---------------------------------------------------------------------------
__global__ void norm_q_kernel(const float* __restrict__ qf,
                              unsigned short* __restrict__ Qbf) {
    int wid  = (blockIdx.x * blockDim.x + threadIdx.x) >> 5;
    int lane = threadIdx.x & 31;
    if (wid >= T * WQ * C) return;
    int c  = wid % C;
    int tq = wid / C;

    const float* src = qf + ((size_t)tq * C + c) * HW;
    float v[4];
    float ss = 0.f;
#pragma unroll
    for (int i = 0; i < 4; ++i) {
        int x = lane + 32 * i;
        v[i] = (x < HW) ? src[x] : 0.f;
        ss += v[i] * v[i];
    }
#pragma unroll
    for (int off = 16; off >= 1; off >>= 1) ss += __shfl_xor(ss, off, 32);
    float scale = 1.0f / fmaxf(sqrtf(ss), EPS);

    unsigned short* dst = Qbf + (size_t)tq * MPAD * C + c;   // stride C per x
#pragma unroll
    for (int i = 0; i < 4; ++i) {
        int x = lane + 32 * i;
        if (x < HW)        dst[(size_t)x * C] = f32_to_bf16(v[i] * scale);
        else if (x < MPAD) dst[(size_t)x * C] = 0;
    }
}

// ---------------------------------------------------------------------------
// Kernel 2: normalize support over c (stride-100 gather), convert to bf16,
// write B-layout Sbf[(t*5+w)][y(0..511)][c]; rows 500..511 = 0.
// One wave32 per (t,way,y).
// ---------------------------------------------------------------------------
__global__ void norm_s_kernel(const float* __restrict__ sf,
                              unsigned short* __restrict__ Sbf) {
    int wid  = (blockIdx.x * blockDim.x + threadIdx.x) >> 5;
    int lane = threadIdx.x & 31;
    if (wid >= T * WAY * NPAD) return;
    int y   = wid % NPAD;
    int tw  = wid / NPAD;          // t*WAY + way
    int way = tw % WAY;
    int t   = tw / WAY;

    unsigned short* dst = Sbf + ((size_t)tw * NPAD + y) * C;
    if (y >= NREAL) {              // uniform per wave: zero padding row
#pragma unroll
        for (int i = 0; i < C / 32; ++i) dst[lane + 32 * i] = 0;
        return;
    }
    int shot = y / HW;
    int x    = y % HW;
    const float* src = sf + ((size_t)((t * WAY + way) * SHOT + shot) * C) * HW + x;

    float vals[C / 32];
    float ss = 0.f;
#pragma unroll
    for (int i = 0; i < C / 32; ++i) {
        int cc = lane + 32 * i;
        float v = src[(size_t)cc * HW];
        vals[i] = v;
        ss += v * v;
    }
#pragma unroll
    for (int off = 16; off >= 1; off >>= 1) ss += __shfl_xor(ss, off, 32);
    float scale = 1.0f / fmaxf(sqrtf(ss), EPS);
#pragma unroll
    for (int i = 0; i < C / 32; ++i)
        dst[lane + 32 * i] = f32_to_bf16(vals[i] * scale);
}

// ---------------------------------------------------------------------------
// Kernel 3: fused GEMM (bf16 WMMA, fp32 acc) + streaming top-3 + row sum.
// grid = 1500 blocks, one per (t,q,way). 224 threads = 7 waves, wave = M tile.
// B is staged per 64-column batch into LDS ONCE per block via
// global_load_async_to_lds_b128 (ASYNCcnt) and shared by all 7 waves;
// K = 640 -> 20 x v_wmma_f32_16x16x32_bf16 per 16x16 tile.
// ---------------------------------------------------------------------------
__global__ void __launch_bounds__(224)
dn4_gemm_topk_kernel(const unsigned short* __restrict__ Qbf,
                     const unsigned short* __restrict__ Sbf,
                     float* __restrict__ out) {
    const int g     = blockIdx.x;            // ((t*75+q)*5 + w)
    const int w     = g % WAY;
    const int tq    = g / WAY;               // t*75 + q
    const int t     = g / (WQ * WAY);
    const int mtile = threadIdx.x >> 5;      // 0..6
    const int lane  = threadIdx.x & 31;
    const int hi    = lane >> 4;             // 0 or 1
    const int lr    = lane & 15;

    __shared__ unsigned short sB[NB * LDSROW];   // 64 x 648 bf16 = ~81 KB
    __shared__ float sPart[14];

    const unsigned short* Abase = Qbf + (size_t)tq * MPAD * C;
    const unsigned short* Bbase = Sbf + (size_t)(t * WAY + w) * NPAD * C;
    const unsigned short* Arow  = Abase + (size_t)(mtile * 16 + lr) * C;

    // per-lane streaming top-3 per C-row register r (descending)
    float h0[8], h1[8], h2[8];
#pragma unroll
    for (int r = 0; r < 8; ++r) { h0[r] = -3.402823466e38f; h1[r] = h0[r]; h2[r] = h0[r]; }

    constexpr int CHUNKS_PER_ROW = C / 8;              // 80 x 16B per row
    constexpr int TOT_CHUNKS     = NB * CHUNKS_PER_ROW; // 5120

    for (int ntb = 0; ntb < 8; ++ntb) {      // 8 batches of 4 N-tiles
        if (ntb) __syncthreads();            // previous batch fully consumed

        // ---- cooperative async stage of B[ntb*64 .. +63][0..639] into LDS ----
        for (int i = threadIdx.x; i < TOT_CHUNKS; i += 224) {
            const int row = i / CHUNKS_PER_ROW;
            const int ch  = i % CHUNKS_PER_ROW;
            const unsigned short* gp =
                Bbase + (size_t)(ntb * NB + row) * C + ch * 8;
            const unsigned ldsoff =
                (unsigned)(size_t)(&sB[row * LDSROW + ch * 8]);
            asm volatile("global_load_async_to_lds_b128 %0, %1, off"
                         :: "v"(ldsoff), "v"((unsigned long long)gp)
                         : "memory");
        }
        asm volatile("s_wait_asynccnt 0x0" ::: "memory");
        __syncthreads();                     // staged data visible to all waves

        v8f acc[4];
#pragma unroll
        for (int j = 0; j < 4; ++j) acc[j] = {};

        for (int ks = 0; ks < C / 32; ++ks) {      // 20 K slabs
            const int ka = ks * 32 + hi * 8;       // A: lanes0-15 K{0..7,16..23}, lanes16-31 K{8..15,24..31}
            Frag a;
            a.u[0] = *(const uint4*)(Arow + ka);
            a.u[1] = *(const uint4*)(Arow + ka + 16);
            const int kb = ks * 32 + hi * 8;       // B: lanes0-15 K{0..7,16..23}, lanes16-31 K{8..15,24..31} for col n=lr
#pragma unroll
            for (int j = 0; j < 4; ++j) {
                const unsigned short* brow = &sB[(j * 16 + lr) * LDSROW + kb];
                Frag b;
                b.u[0] = *(const uint4*)(brow);
                b.u[1] = *(const uint4*)(brow + 16);
                acc[j] = __builtin_amdgcn_wmma_f32_16x16x32_bf16(
                    /*neg_a=*/false, a.v, /*neg_b=*/false, b.v,
                    /*c_mod=*/(short)0, acc[j],
                    /*reuse_a=*/false, /*reuse_b=*/false);
            }
        }

        // streaming top-3 update; mask padded columns y >= 500
#pragma unroll
        for (int j = 0; j < 4; ++j) {
            const int n = (ntb * 4 + j) * 16 + lr;
            if (n < NREAL) {
#pragma unroll
                for (int r = 0; r < 8; ++r) ins3(acc[j][r], h0[r], h1[r], h2[r]);
            }
        }
    }

    // merge top-3 across the 16 lanes sharing each row (xor offsets stay in half)
#pragma unroll
    for (int r = 0; r < 8; ++r) {
#pragma unroll
        for (int off = 1; off < 16; off <<= 1) {
            float o0 = __shfl_xor(h0[r], off, 32);
            float o1 = __shfl_xor(h1[r], off, 32);
            float o2 = __shfl_xor(h2[r], off, 32);
            ins3(o0, h0[r], h1[r], h2[r]);
            ins3(o1, h0[r], h1[r], h2[r]);
            ins3(o2, h0[r], h1[r], h2[r]);
        }
    }

    // sum top-3 over valid rows (x < 100); C layout: row = mtile*16 + hi*8 + r
    float part = 0.f;
#pragma unroll
    for (int r = 0; r < 8; ++r) {
        int row = mtile * 16 + hi * 8 + r;
        if (row < HW) part += h0[r] + h1[r] + h2[r];
    }

    if (lr == 0) sPart[mtile * 2 + hi] = part;   // deterministic slot writes
    __syncthreads();
    if (threadIdx.x == 0) {
        float s = 0.f;
        for (int i = 0; i < 14; ++i) s += sPart[i];
        out[g] = s;
    }
}

// ---------------------------------------------------------------------------
extern "C" void kernel_launch(void* const* d_in, const int* in_sizes, int n_in,
                              void* d_out, int out_size, void* d_ws, size_t ws_size,
                              hipStream_t stream) {
    const float* qf = (const float*)d_in[0];
    const float* sf = (const float*)d_in[1];
    float* out = (float*)d_out;

    const size_t qbf_bytes = (size_t)T * WQ * MPAD * C * sizeof(unsigned short); // ~43.0 MB
    unsigned short* Qbf = (unsigned short*)d_ws;
    unsigned short* Sbf = (unsigned short*)((char*)d_ws + qbf_bytes);            // ~13.1 MB

    {   // one wave per (t,q,c) vector
        const int waves = T * WQ * C;                 // 192000
        const int threads = 256;
        const int blocks = (waves * 32 + threads - 1) / threads;
        norm_q_kernel<<<blocks, threads, 0, stream>>>(qf, Qbf);
    }
    {   // one wave per (t,way,y)
        const int waves = T * WAY * NPAD;             // 10240
        const int threads = 256;
        const int blocks = (waves * 32 + threads - 1) / threads;
        norm_s_kernel<<<blocks, threads, 0, stream>>>(sf, Sbf);
    }
    dn4_gemm_topk_kernel<<<T * WQ * WAY, 224, 0, stream>>>(Qbf, Sbf, out);
}
